// TemporalDifferentModuleMSDeformAttnVIZ_34196529610807
// MI455X (gfx1250) — compile-verified, matrix operand order checked by reading, not run on hardware
//
#include <hip/hip_runtime.h>

typedef __attribute__((ext_vector_type(16))) _Float16 v16h;
typedef __attribute__((ext_vector_type(8)))  float    v8f;

#define LQ 16320

// Per-level constants (SHAPES = {T,H,W}, frames stacked vertically -> grid (T*H, W))
__constant__ int c_w[4]     = {64, 32, 16, 8};
__constant__ int c_th[4]    = {192, 96, 48, 24};     // T*H
__constant__ int c_start[4] = {0, 12288, 15360, 16128};

// ---------------------------------------------------------------------------
// Pack f32 activation matrix A[M,K] into f16 WMMA A-fragments.
// Fragment (mtile, kbi): lane l holds 16 halves; rows lr=l&15,
// K = kbi*32 + (l>>4)*8 + {0..7, 16..23}.  One wave per fragment.
// ---------------------------------------------------------------------------
__global__ __launch_bounds__(256)
void pack_a_f16(const float* __restrict__ A, v16h* __restrict__ P, int M, int K) {
    const int KB   = K >> 5;
    const int frag = blockIdx.x * 8 + (threadIdx.x >> 5);
    if (frag >= (M >> 4) * KB) return;
    const int mtile = frag / KB, kbi = frag % KB;
    const int lane = threadIdx.x & 31;
    const int lr = lane & 15, half = lane >> 4;

    const float* arow = A + (size_t)(mtile * 16 + lr) * K + kbi * 32 + half * 8;
    const float4 a0 = *(const float4*)(arow + 0);
    const float4 a1 = *(const float4*)(arow + 4);
    const float4 a2 = *(const float4*)(arow + 16);
    const float4 a3 = *(const float4*)(arow + 20);

    v16h v;
    v[0] = (_Float16)a0.x;  v[1] = (_Float16)a0.y;  v[2]  = (_Float16)a0.z;  v[3]  = (_Float16)a0.w;
    v[4] = (_Float16)a1.x;  v[5] = (_Float16)a1.y;  v[6]  = (_Float16)a1.z;  v[7]  = (_Float16)a1.w;
    v[8] = (_Float16)a2.x;  v[9] = (_Float16)a2.y;  v[10] = (_Float16)a2.z;  v[11] = (_Float16)a2.w;
    v[12]= (_Float16)a3.x;  v[13]= (_Float16)a3.y;  v[14] = (_Float16)a3.z;  v[15] = (_Float16)a3.w;
    P[(size_t)frag * 32 + lane] = v;
}

// ---------------------------------------------------------------------------
// Pack f32 weight matrix W[K,N] into f16 WMMA B-fragments (strided reads done once).
// Fragment (ntile, kbi): lane l holds col nc=l&15, same K packing as A.
// ---------------------------------------------------------------------------
__global__ __launch_bounds__(256)
void pack_b_f16(const float* __restrict__ W, v16h* __restrict__ P, int N, int K) {
    const int KB   = K >> 5;
    const int frag = blockIdx.x * 8 + (threadIdx.x >> 5);
    if (frag >= (N >> 4) * KB) return;
    const int ntile = frag / KB, kbi = frag % KB;
    const int lane = threadIdx.x & 31;
    const int nc = lane & 15, half = lane >> 4;

    const float* wcol = W + ntile * 16 + nc;
    const int k0 = kbi * 32 + half * 8;
    v16h v;
#pragma unroll
    for (int e = 0; e < 8; ++e) {
        v[e]     = (_Float16)wcol[(size_t)(k0 + e) * N];
        v[e + 8] = (_Float16)wcol[(size_t)(k0 + 16 + e) * N];
    }
    P[(size_t)frag * 32 + lane] = v;
}

// ---------------------------------------------------------------------------
// WMMA GEMM on pre-packed fragments: C[M,N] = A*W + bias.
// One wave computes a 16x64 C strip (1 A-frag reused by 4 N-tiles per k-step):
// per k-step = 5 x (2xb128 loads) -> 4 x v_wmma_f32_16x16x32_f16.
// ---------------------------------------------------------------------------
__global__ __launch_bounds__(128)
void gemm_wmma_packed(const v16h* __restrict__ PA, const v16h* __restrict__ PB,
                      const float* __restrict__ bias, float* __restrict__ C,
                      int M, int N, int K) {
    const int lane  = threadIdx.x & 31;
    const int wave  = threadIdx.x >> 5;
    const int mtile = blockIdx.x * 4 + wave;
    const int ng    = blockIdx.y;          // group of 4 N-tiles (64 cols)
    const int KB    = K >> 5;
    if (mtile * 16 >= M) return;

    const v16h* pa = PA + (size_t)mtile * KB * 32 + lane;
    const v16h* pb = PB + (size_t)(ng * 4) * KB * 32 + lane;

    v8f acc[4] = {};
    for (int kbi = 0; kbi < KB; ++kbi) {
        const v16h a = pa[(size_t)kbi * 32];
        __builtin_prefetch(pa + (size_t)(kbi + 1) * 32, 0, 0);  // global_prefetch_b8
        v16h bfr[4];
#pragma unroll
        for (int nt = 0; nt < 4; ++nt)
            bfr[nt] = pb[((size_t)nt * KB + kbi) * 32];
#pragma unroll
        for (int nt = 0; nt < 4; ++nt)
            acc[nt] = __builtin_amdgcn_wmma_f32_16x16x32_f16(
                false, a, false, bfr[nt], (short)0, acc[nt], false, false);
    }

    const int lr = lane & 15, half = lane >> 4;
#pragma unroll
    for (int nt = 0; nt < 4; ++nt) {
        const int col = ng * 64 + nt * 16 + lr;
        const float bv = bias[col];
#pragma unroll
        for (int r = 0; r < 8; ++r) {
            const int row = mtile * 16 + r + half * 8;   // C/D: M = vgpr + 8*(lane>=16)
            C[(size_t)row * N + col] = acc[nt][r] + bv;
        }
    }
}

// ---------------------------------------------------------------------------
// Fused joint-softmax + sampling-location + bilinear deformable gather.
// One wave32 per (query, head): lane = value channel (DH=32) AND lane = one of
// the 32 softmax logits. Corner gathers are single coalesced 128B loads.
// ---------------------------------------------------------------------------
__global__ __launch_bounds__(256)
void deform_sample(const float* __restrict__ value,  // (LQ, NH*DH)
                   const float* __restrict__ ref,    // (LQ, L, 2)
                   const float* __restrict__ toff,   // (LQ, L, TW, 2)
                   const float* __restrict__ cso,    // (LQ, 256)
                   const float* __restrict__ tso,    // (LQ, 256)
                   const float* __restrict__ caw,    // (LQ, 128)
                   const float* __restrict__ taw,    // (LQ, 128)
                   float* __restrict__ samp)         // (LQ, 256)
{
    const int q    = blockIdx.x;
    const int h    = threadIdx.x >> 5;
    const int lane = threadIdx.x & 31;

    // ---- joint softmax over [caw(16) | taw(16)] ; lane i owns logit i ----
    float logit = (lane < 16) ? caw[(size_t)q * 128 + h * 16 + lane]
                              : taw[(size_t)q * 128 + h * 16 + (lane - 16)];
    float m = logit;
#pragma unroll
    for (int off = 16; off; off >>= 1) m = fmaxf(m, __shfl_xor(m, off, 32));
    float e = __expf(logit - m);
    float s = e;
#pragma unroll
    for (int off = 16; off; off >>= 1) s += __shfl_xor(s, off, 32);
    const float wgt = e / s;

    // ---- each lane computes the sampling location of its own point ----
    float lx, ly;
    if (lane < 16) {                       // current points: l = lane>>2, p = lane&3
        const int li = lane >> 2, p = lane & 3;
        const int base = ((h * 4 + li) * 4 + p) * 2;
        const float ox = cso[(size_t)q * 256 + base];
        const float oy = cso[(size_t)q * 256 + base + 1];
        lx = ref[(size_t)q * 8 + li * 2 + 0] + ox / (float)c_w[li];
        ly = ref[(size_t)q * 8 + li * 2 + 1] + oy / (float)c_th[li];
    } else {                               // temporal points: j = tw*NTP + tp
        const int j16 = lane - 16;
        const int li = j16 >> 2, j = j16 & 3, tw = j >> 1, tp = j & 1;
        const int base = (((h * 4 + li) * 2 + tw) * 2 + tp) * 2;
        const float ox = tso[(size_t)q * 256 + base];
        const float oy = tso[(size_t)q * 256 + base + 1];
        const float tx = toff[(size_t)q * 16 + li * 4 + tw * 2 + 0];
        const float ty = toff[(size_t)q * 16 + li * 4 + tw * 2 + 1];
        lx = ref[(size_t)q * 8 + li * 2 + 0] + tx + ox / (float)c_w[li];
        ly = ref[(size_t)q * 8 + li * 2 + 1] + ty + oy / (float)c_th[li];
    }

    // ---- bilinear gather-accumulate: lane = channel d ----
    const float* vh = value + h * 32 + lane;
    float acc = 0.0f;
#pragma unroll 4
    for (int pi = 0; pi < 32; ++pi) {
        const int l = pi >> 3, pp = pi & 7;
        // attn order per level: p<4 -> caw lane l*4+p ; p>=4 -> taw lane 16+l*4+(p-4)
        const int wl = (pp < 4) ? (l * 4 + pp) : (16 + l * 4 + (pp - 4));
        const float plx = __shfl(lx, wl, 32);
        const float ply = __shfl(ly, wl, 32);
        const float pw  = __shfl(wgt, wl, 32);

        const int Wl = c_w[l], THl = c_th[l], st = c_start[l];
        const float x = plx * (float)Wl - 0.5f;
        const float y = ply * (float)THl - 0.5f;
        const float xf = floorf(x), yf = floorf(y);
        const float fx = x - xf, fy = y - yf;
        const int x0 = (int)xf, y0 = (int)yf;

        float v00 = 0.f, v01 = 0.f, v10 = 0.f, v11 = 0.f;
        const bool xi0 = (x0 >= 0) & (x0 < Wl),   xi1 = (x0 + 1 >= 0) & (x0 + 1 < Wl);
        const bool yi0 = (y0 >= 0) & (y0 < THl),  yi1 = (y0 + 1 >= 0) & (y0 + 1 < THl);
        if (xi0 & yi0) v00 = vh[(size_t)(st + y0 * Wl + x0) * 256];
        if (xi1 & yi0) v01 = vh[(size_t)(st + y0 * Wl + x0 + 1) * 256];
        if (xi0 & yi1) v10 = vh[(size_t)(st + (y0 + 1) * Wl + x0) * 256];
        if (xi1 & yi1) v11 = vh[(size_t)(st + (y0 + 1) * Wl + x0 + 1) * 256];

        const float bil = v00 * (1.f - fx) * (1.f - fy) + v01 * fx * (1.f - fy)
                        + v10 * (1.f - fx) * fy         + v11 * fx * fy;
        acc += pw * bil;
    }
    samp[(size_t)q * 256 + h * 32 + lane] = acc;
}

// ---------------------------------------------------------------------------
extern "C" void kernel_launch(void* const* d_in, const int* in_sizes, int n_in,
                              void* d_out, int out_size, void* d_ws, size_t ws_size,
                              hipStream_t stream) {
    const float* query = (const float*)d_in[0];
    const float* ref   = (const float*)d_in[1];
    const float* toff  = (const float*)d_in[2];
    const float* inpf  = (const float*)d_in[3];
    // d_in[4], d_in[5]: spatial shapes / level start index (compile-time constants here)
    const float* W_so  = (const float*)d_in[6];
    const float* b_so  = (const float*)d_in[7];
    const float* W_tso = (const float*)d_in[8];
    const float* b_tso = (const float*)d_in[9];
    const float* W_aw  = (const float*)d_in[10];
    const float* b_aw  = (const float*)d_in[11];
    const float* W_taw = (const float*)d_in[12];
    const float* b_taw = (const float*)d_in[13];
    const float* W_v   = (const float*)d_in[14];
    const float* b_v   = (const float*)d_in[15];
    const float* W_o   = (const float*)d_in[16];
    const float* b_o   = (const float*)d_in[17];

    // ---- workspace carve-up (all offsets 32B-aligned) ----
    char* p = (char*)d_ws;
    float* value = (float*)p;  p += (size_t)LQ * 256 * 4;
    float* cso   = (float*)p;  p += (size_t)LQ * 256 * 4;
    float* tso   = (float*)p;  p += (size_t)LQ * 256 * 4;
    float* cawb  = (float*)p;  p += (size_t)LQ * 128 * 4;
    float* tawb  = (float*)p;  p += (size_t)LQ * 128 * 4;
    float* samp  = (float*)p;  p += (size_t)LQ * 256 * 4;
    v16h* pq     = (v16h*)p;   p += (size_t)LQ * 256 * 2;   // packed query (f16)
    v16h* pin    = (v16h*)p;   p += (size_t)LQ * 256 * 2;   // packed input_flatten
    v16h* psamp  = (v16h*)p;   p += (size_t)LQ * 256 * 2;   // packed sampled output
    v16h* pWv    = (v16h*)p;   p += 256 * 256 * 2;
    v16h* pWso   = (v16h*)p;   p += 256 * 256 * 2;
    v16h* pWtso  = (v16h*)p;   p += 256 * 256 * 2;
    v16h* pWo    = (v16h*)p;   p += 256 * 256 * 2;
    v16h* pWaw   = (v16h*)p;   p += 256 * 128 * 2;
    v16h* pWtaw  = (v16h*)p;   p += 256 * 128 * 2;

    const dim3 blk256(256), blk128(128);

    // ---- pack weights (once; strided reads happen only here) ----
    pack_b_f16<<<16, blk256, 0, stream>>>(W_v,   pWv,   256, 256);  // 128 frags
    pack_b_f16<<<16, blk256, 0, stream>>>(W_so,  pWso,  256, 256);
    pack_b_f16<<<16, blk256, 0, stream>>>(W_tso, pWtso, 256, 256);
    pack_b_f16<<<16, blk256, 0, stream>>>(W_o,   pWo,   256, 256);
    pack_b_f16<<< 8, blk256, 0, stream>>>(W_aw,  pWaw,  128, 256);  // 64 frags
    pack_b_f16<<< 8, blk256, 0, stream>>>(W_taw, pWtaw, 128, 256);

    // ---- pack activations (f32 -> f16 fragments, b128 loads) ----
    pack_a_f16<<<1020, blk256, 0, stream>>>(query, pq,  LQ, 256);   // 8160 frags
    pack_a_f16<<<1020, blk256, 0, stream>>>(inpf,  pin, LQ, 256);

    // ---- projection GEMMs: M=16320 -> 255 blocks x 4 waves; N/64 tile-groups ----
    gemm_wmma_packed<<<dim3(255, 4), blk128, 0, stream>>>(pin, pWv,   b_v,   value, LQ, 256, 256);
    gemm_wmma_packed<<<dim3(255, 4), blk128, 0, stream>>>(pq,  pWso,  b_so,  cso,   LQ, 256, 256);
    gemm_wmma_packed<<<dim3(255, 4), blk128, 0, stream>>>(pq,  pWtso, b_tso, tso,   LQ, 256, 256);
    gemm_wmma_packed<<<dim3(255, 2), blk128, 0, stream>>>(pq,  pWaw,  b_aw,  cawb,  LQ, 128, 256);
    gemm_wmma_packed<<<dim3(255, 2), blk128, 0, stream>>>(pq,  pWtaw, b_taw, tawb,  LQ, 128, 256);

    // ---- fused softmax + deformable bilinear sampling ----
    deform_sample<<<LQ, blk256, 0, stream>>>(value, ref, toff, cso, tso, cawb, tawb, samp);

    // ---- output projection ----
    pack_a_f16<<<1020, blk256, 0, stream>>>(samp, psamp, LQ, 256);
    gemm_wmma_packed<<<dim3(255, 4), blk128, 0, stream>>>(psamp, pWo, b_o, (float*)d_out, LQ, 256, 256);
}
